// FullCovarianceModule_79723182948721
// MI455X (gfx1250) — matrix operand order repaired and checked by reading, not compiled, not run on hardware
//
#include <hip/hip_runtime.h>

typedef __attribute__((ext_vector_type(16))) __bf16 v16bf;
typedef __attribute__((ext_vector_type(8)))  __bf16 v8bf;
typedef __attribute__((ext_vector_type(4)))  __bf16 v4bf;
typedef __attribute__((ext_vector_type(8)))  float  v8f;
typedef __attribute__((ext_vector_type(4)))  unsigned int u32x4;
typedef __attribute__((ext_vector_type(8)))  int i32x8;
typedef __attribute__((ext_vector_type(4)))  int i32x4;

#if defined(__has_builtin)
#if __has_builtin(__builtin_amdgcn_tensor_load_to_lds)
#define HAVE_TDM 1
#endif
#endif

// ---------------------------------------------------------------------------
// WMMA fragment builders (CDNA5 16x16x32 bf16 layouts, ISA 7.12.2)
// A (16xK=32): lane m (0-15) holds K={kb..kb+7, kb+16..kb+23}, kb = 8*(lane>=16)
// B (K=32x16): lane n (0-15) holds K={kb..kb+7, kb+16..kb+23} of column n, kb = 8*(lane>=16)
// ---------------------------------------------------------------------------
__device__ inline v16bf frag_a(const __bf16* base) {
  v8bf c0 = *(const v8bf*)(base);
  v8bf c1 = *(const v8bf*)(base + 16);
  v16bf r;
#pragma unroll
  for (int i = 0; i < 8; ++i) { r[i] = c0[i]; r[i + 8] = c1[i]; }
  return r;
}
__device__ inline v16bf frag_b(const __bf16* base) {
  v8bf c0 = *(const v8bf*)(base);
  v8bf c1 = *(const v8bf*)(base + 16);
  v16bf r;
#pragma unroll
  for (int i = 0; i < 8; ++i) { r[i] = c0[i]; r[i + 8] = c1[i]; }
  return r;
}
__device__ inline v8f wmma_bf16(v16bf a, v16bf b, v8f c) {
  return __builtin_amdgcn_wmma_f32_16x16x32_bf16(false, a, false, b, (short)0, c,
                                                 false, false);
}
__device__ inline void split_bf16(float v, __bf16& h, __bf16& l) {
  h = (__bf16)v;
  l = (__bf16)(v - (float)h);
}
__device__ inline void cvt4(float4 v, int o, v16bf& h, v16bf& l) {
  __bf16 hh, ll;
  split_bf16(v.x, hh, ll); h[o + 0] = hh; l[o + 0] = ll;
  split_bf16(v.y, hh, ll); h[o + 1] = hh; l[o + 1] = ll;
  split_bf16(v.z, hh, ll); h[o + 2] = hh; l[o + 2] = ll;
  split_bf16(v.w, hh, ll); h[o + 3] = hh; l[o + 3] = ll;
}

__device__ inline unsigned lds_offset(const void* p) {
  // generic pointer to __shared__: low 32 bits hold the LDS byte offset
  return (unsigned)(unsigned long long)p;
}

#ifdef HAVE_TDM
// ---------------------------------------------------------------------------
// TDM: 2D tile load global->LDS. tile0 = row length (elements, 2B each),
// tile1 = #rows, stride0 = global row stride (elements). LDS rows padded:
// pad_interval=3 (every 16 DWORDs = 64B = one 32-elem row), pad_amount=3
// (4 DWORDs = 16B) -> effective LDS row stride = 40 bf16 elements.
// D# layout per CDNA5 ISA 8.3/8.4: count=1, type=2, data_size=1 (2 bytes).
// ---------------------------------------------------------------------------
__device__ inline void tdm_load_2d_bf16(unsigned lds_off, const void* gptr,
                                        int tile0, int tile1, int stride0) {
  unsigned long long ga = (unsigned long long)gptr;
  u32x4 g0;
  g0[0] = 1u;                                   // count=1, user mode
  g0[1] = lds_off;                              // lds_addr
  g0[2] = (unsigned)(ga & 0xFFFFFFFFu);         // global_addr[31:0]
  g0[3] = (unsigned)((ga >> 32) & 0x01FFFFFFu) | (2u << 30);  // [56:32] | type=2
  i32x8 g1;
  g1[0] = (int)((1u << 16)      // data_size = 1 (2 bytes)
              | (1u << 20)      // pad_enable
              | (3u << 22)      // pad_interval: 16 DWORDs
              | (3u << 25));    // pad_amount: 4 DWORDs
  g1[1] = (int)(((unsigned)tile0 & 0xFFFFu) << 16);                  // tensor_dim0 lo
  g1[2] = (int)((((unsigned)tile0 >> 16) & 0xFFFFu) |
                (((unsigned)tile1 & 0xFFFFu) << 16));                // dim0 hi | dim1 lo
  g1[3] = (int)((((unsigned)tile1 >> 16) & 0xFFFFu) |
                (((unsigned)tile0 & 0xFFFFu) << 16));                // dim1 hi | tile_dim0
  g1[4] = (int)((unsigned)tile1 & 0xFFFFu);                          // tile_dim1, tile_dim2=0
  g1[5] = stride0;                                                   // dim0_stride[31:0]
  g1[6] = 0;                                                         // stride hi bits
  g1[7] = 0;
  i32x4 z4 = {0, 0, 0, 0};
#if __clang_major__ >= 23
  i32x8 z8 = {};
  __builtin_amdgcn_tensor_load_to_lds(g0, g1, z4, z4, z8, 0);
#else
  __builtin_amdgcn_tensor_load_to_lds(g0, g1, z4, z4, 0);
#endif
}
#endif

// ---------------------------------------------------------------------------
// Kernel 1: split W[528x512] f32 -> Whi/Wlo bf16 once (W stays L2-resident).
// ---------------------------------------------------------------------------
__global__ __launch_bounds__(256) void wsplit_kernel(const float* __restrict__ W,
                                                     __bf16* __restrict__ Whi,
                                                     __bf16* __restrict__ Wlo) {
  int u = blockIdx.x * 256 + threadIdx.x;  // 0..16895, 16 floats each
  const float4* src = (const float4*)W + (size_t)u * 4;
#pragma unroll
  for (int j = 0; j < 4; ++j) {
    float4 v = src[j];
    __bf16 h0, h1, h2, h3, l0, l1, l2, l3;
    split_bf16(v.x, h0, l0); split_bf16(v.y, h1, l1);
    split_bf16(v.z, h2, l2); split_bf16(v.w, h3, l3);
    v4bf hv = {h0, h1, h2, h3};
    v4bf lv = {l0, l1, l2, l3};
    *(v4bf*)(Whi + (size_t)u * 16 + j * 4) = hv;
    *(v4bf*)(Wlo + (size_t)u * 16 + j * 4) = lv;
  }
}

// ---------------------------------------------------------------------------
// Kernel 2: y[N,528] = x[N,512] @ W^T + b, split-bf16 compensated WMMA.
// 128x176 tile per WG (528 = 3*176), 8 waves; wave w owns row tile w and 11
// accumulator tiles. x staged via VALU split; W tile staged via TDM (wave 0
// issues tensor_load_to_lds for hi and lo, overlapping the x staging).
// ---------------------------------------------------------------------------
#define APAD 40
#define BPAD 40
__global__ __launch_bounds__(256) void gemm_kernel(const float* __restrict__ x,
                                                   const __bf16* __restrict__ Whi,
                                                   const __bf16* __restrict__ Wlo,
                                                   const float* __restrict__ bias,
                                                   float* __restrict__ y) {
  __shared__ __bf16 Ah[128 * APAD];
  __shared__ __bf16 Al[128 * APAD];
  __shared__ __bf16 Bh[176 * BPAD];
  __shared__ __bf16 Bl[176 * BPAD];

  const int tid = threadIdx.x;
  const int lane = tid & 31;
  const int wave = tid >> 5;
  const int row0 = blockIdx.x * 128;
  const int n0 = blockIdx.y * 176;
  const int hiHalf = (lane >> 4) & 1;
  const int l15 = lane & 15;

  v8f acc[11];
#pragma unroll
  for (int t = 0; t < 11; ++t) acc[t] = (v8f){};

  for (int k0 = 0; k0 < 512; k0 += 32) {
#ifdef HAVE_TDM
    if (wave == 0) {  // async DMA of W tile (176 rows x 32 bf16, padded rows)
      tdm_load_2d_bf16(lds_offset(Bh), Whi + (size_t)n0 * 512 + k0, 32, 176, 512);
      tdm_load_2d_bf16(lds_offset(Bl), Wlo + (size_t)n0 * 512 + k0, 32, 176, 512);
    }
#else
    for (int u = tid; u < 176 * 8; u += 256) {
      int col = u >> 3, rem = u & 7, sel = rem >> 2, ch = rem & 3;
      const __bf16* src = (sel ? Wlo : Whi) + (size_t)(n0 + col) * 512 + k0 + ch * 8;
      __bf16* dst = (sel ? Bl : Bh) + col * BPAD + ch * 8;
      *(v8bf*)dst = *(const v8bf*)src;
    }
#endif
    // --- stage A: x[128][k0..k0+32) f32 -> bf16 hi/lo in LDS (overlaps TDM) ---
#pragma unroll
    for (int j = 0; j < 4; ++j) {
      int u = tid + 256 * j;  // 0..1023
      int r = u >> 3;         // row 0..127
      int f4 = u & 7;         // float4 index within the 32-float k chunk
      const float* gp = x + (size_t)(row0 + r) * 512 + k0 + f4 * 4;
      if (k0 + 32 < 512) __builtin_prefetch(gp + 32, 0, 0);  // next k-step
      float4 v = *(const float4*)gp;
      __bf16 h0, h1, h2, h3, l0, l1, l2, l3;
      split_bf16(v.x, h0, l0); split_bf16(v.y, h1, l1);
      split_bf16(v.z, h2, l2); split_bf16(v.w, h3, l3);
      v4bf hv = {h0, h1, h2, h3};
      v4bf lv = {l0, l1, l2, l3};
      *(v4bf*)(Ah + r * APAD + f4 * 4) = hv;
      *(v4bf*)(Al + r * APAD + f4 * 4) = lv;
    }
#ifdef HAVE_TDM
    if (wave == 0) __builtin_amdgcn_s_wait_tensorcnt(0);
#endif
    __syncthreads();

    v16bf a_hi = frag_a(Ah + (wave * 16 + l15) * APAD + hiHalf * 8);
    v16bf a_lo = frag_a(Al + (wave * 16 + l15) * APAD + hiHalf * 8);
#pragma unroll
    for (int t = 0; t < 11; ++t) {
      v16bf b_hi = frag_b(Bh + (t * 16 + l15) * BPAD + hiHalf * 8);
      v16bf b_lo = frag_b(Bl + (t * 16 + l15) * BPAD + hiHalf * 8);
      acc[t] = wmma_bf16(a_hi, b_hi, acc[t]);  // main product
      acc[t] = wmma_bf16(a_hi, b_lo, acc[t]);  // compensation terms
      acc[t] = wmma_bf16(a_lo, b_hi, acc[t]);
    }
    __syncthreads();
  }

  // epilogue: + bias, store y. C layout: VGPR i -> M = i + 8*hiHalf, N = lane&15
#pragma unroll
  for (int t = 0; t < 11; ++t) {
    int col = n0 + t * 16 + l15;
    float bv = bias[col];
#pragma unroll
    for (int i = 0; i < 8; ++i) {
      int m = i + hiHalf * 8;
      y[(size_t)(row0 + wave * 16 + m) * 528 + col] = acc[t][i] + bv;
    }
  }
}

// ---------------------------------------------------------------------------
// Kernel 3: cov[s] = L_s L_s^T + 1e-4 I.  8 samples per WG. y is scattered
// into a zeroed f32 L[8][32][44] in LDS using per-lane async global->LDS
// copies (ASYNCcnt), then each 16x16 cov tile is one K=32 WMMA (x3 split).
// ---------------------------------------------------------------------------
#define SPW 8
#define LPADF 44
__global__ __launch_bounds__(256) void cov_kernel(const float* __restrict__ y,
                                                  float* __restrict__ out) {
  __shared__ float Lf[SPW * 32 * LPADF];
  const int tid = threadIdx.x;
  const int lane = tid & 31;
  const int wave = tid >> 5;
  const int s0 = blockIdx.x * SPW;

  {  // zero L (upper triangle + pad must be 0 for the K=32 WMMA)
    uint4 z = {0u, 0u, 0u, 0u};
    uint4* p = (uint4*)Lf;
    const int n16 = (SPW * 32 * LPADF * 4) / 16;  // 2816 uint4
    for (int i = tid; i < n16; i += 256) p[i] = z;
  }
  __syncthreads();

  {  // tril scatter via async global->LDS b32 (per-lane LDS dest addresses)
    int s = tid >> 5;
    int l = tid & 31;
    for (int t = l; t < 528; t += 32) {
      int r = (int)((sqrtf(8.0f * (float)t + 1.0f) - 1.0f) * 0.5f);
      if ((r + 1) * (r + 2) / 2 <= t) ++r;
      if (r * (r + 1) / 2 > t) --r;
      int c = t - r * (r + 1) / 2;
      unsigned dst = lds_offset(Lf + (s * 32 + r) * LPADF + c);
      unsigned long long src =
          (unsigned long long)(y + (size_t)(s0 + s) * 528 + t);
      asm volatile("global_load_async_to_lds_b32 %0, %1, off"
                   :: "v"(dst), "v"(src)
                   : "memory");
    }
    asm volatile("s_wait_asynccnt 0" ::: "memory");
  }
  __syncthreads();

  const int hiHalf = (lane >> 4) & 1;
  const int l15 = lane & 15;
  for (int job = wave; job < SPW * 4; job += 8) {
    int s = job >> 2;
    int I = (job >> 1) & 1;
    int J = job & 1;
    // A = L[16I.., 0:32]; B column n = row (16J+n) of L  (B = L^T)
    const float* aB = Lf + (s * 32 + I * 16 + l15) * LPADF + hiHalf * 8;
    const float* bB = Lf + (s * 32 + J * 16 + l15) * LPADF + hiHalf * 8;
    v16bf a_hi, a_lo, b_hi, b_lo;
    cvt4(*(const float4*)(aB + 0), 0, a_hi, a_lo);
    cvt4(*(const float4*)(aB + 4), 4, a_hi, a_lo);
    cvt4(*(const float4*)(aB + 16), 8, a_hi, a_lo);
    cvt4(*(const float4*)(aB + 20), 12, a_hi, a_lo);
    cvt4(*(const float4*)(bB + 0), 0, b_hi, b_lo);
    cvt4(*(const float4*)(bB + 4), 4, b_hi, b_lo);
    cvt4(*(const float4*)(bB + 16), 8, b_hi, b_lo);
    cvt4(*(const float4*)(bB + 20), 12, b_hi, b_lo);
    v8f acc = (v8f){};
    acc = wmma_bf16(a_hi, b_hi, acc);
    acc = wmma_bf16(a_hi, b_lo, acc);
    acc = wmma_bf16(a_lo, b_hi, acc);
    float* dst = out + (size_t)(s0 + s) * 1024 + (I * 16) * 32 + J * 16;
#pragma unroll
    for (int i = 0; i < 8; ++i) {
      int m = i + hiHalf * 8;
      float v = acc[i];
      if (I == J && m == l15) v += 1e-4f;  // + OFFSET * I
      dst[m * 32 + l15] = v;
    }
  }
}

// ---------------------------------------------------------------------------
extern "C" void kernel_launch(void* const* d_in, const int* in_sizes, int n_in,
                              void* d_out, int out_size, void* d_ws,
                              size_t ws_size, hipStream_t stream) {
  (void)in_sizes; (void)n_in; (void)out_size; (void)ws_size;
  const float* x = (const float*)d_in[0];  // [65536, 512]
  const float* W = (const float*)d_in[1];  // [528, 512]
  const float* b = (const float*)d_in[2];  // [528]
  float* out = (float*)d_out;              // [65536, 32, 32]

  char* ws = (char*)d_ws;
  const size_t wbytes = (size_t)528 * 512 * sizeof(__bf16);  // 540672
  __bf16* Whi = (__bf16*)ws;
  __bf16* Wlo = (__bf16*)(ws + wbytes);
  float* y = (float*)(ws + 2 * wbytes);  // [65536, 528] f32 (~138.4 MB)

  wsplit_kernel<<<66, 256, 0, stream>>>(W, Whi, Wlo);
  gemm_kernel<<<dim3(512, 3), 256, 0, stream>>>(x, Whi, Wlo, b, y);
  cov_kernel<<<65536 / SPW, 256, 0, stream>>>(y, out);
}
